// AutoregressiveBase_49907519979875
// MI455X (gfx1250) — matrix operand order repaired and checked by reading, not compiled, not run on hardware
//
#include <hip/hip_runtime.h>
#include <hip/hip_bf16.h>

typedef __attribute__((ext_vector_type(16))) __bf16 v16bf;
typedef __attribute__((ext_vector_type(8)))  float  v8f;

#define B_TOT 32768
#define H_T   50
#define F_IN  9
#define HID   256
#define NS    7
#define NC    2
#define DT_C  0.02f

#define W1_STR 40    // padded row stride (bf16) for K=32 matrices (5*16B rows)
#define K_STR  264   // padded row stride (bf16) for K=256 matrices (33*16B rows)
#define WAVES  8
#define ROWS   (WAVES * 16)   // 128 batch rows per block

// column interleave within each 32-wide group: lets a lane's two N-tile outputs
// (16 columns apart) land in adjacent bf16 slots -> packed b32 stores.
__device__ __host__ __forceinline__ int kperm(int k) {
    return (k & ~31) | ((k & 15) << 1) | ((k >> 4) & 1);
}

__device__ __forceinline__ float fast_tanh(float x) {
#if __has_builtin(__builtin_amdgcn_tanhf)
    return __builtin_amdgcn_tanhf(x);           // native v_tanh_f32 (TRANS pipe)
#else
    float e = __expf(2.0f * x);
    return 1.0f - 2.0f / (e + 1.0f);
#endif
}

__device__ __forceinline__ v8f zero8() {
    v8f z;
    #pragma unroll
    for (int r = 0; r < 8; ++r) z[r] = 0.f;
    return z;
}

// A-fragment (16x32 bf16, lane=M, K striped): two ds_load_b128 at +0 / +32B.
__device__ __forceinline__ v16bf ldA(const __bf16* p) {
    union { v16bf v; float4 f[2]; } u;
    u.f[0] = *(const float4*)(p);
    u.f[1] = *(const float4*)(p + 16);
    return u.v;
}
// B-fragment (32x16 bf16, lane=N, K striped like A): two b128 loads at +0 / +32B.
__device__ __forceinline__ v16bf ldB(const __bf16* p) {
    union { v16bf v; float4 f[2]; } u;
    u.f[0] = *(const float4*)(p);
    u.f[1] = *(const float4*)(p + 16);
    return u.v;
}

// pack two f32 -> bf16 pair (v_cvt_pk_bf16_f32) and one ds_store_b32
__device__ __forceinline__ void store_pk(__bf16* p, float lo, float hi) {
    union { __bf16 h[2]; unsigned u; } pk;
    pk.h[0] = (__bf16)lo;
    pk.h[1] = (__bf16)hi;
    *reinterpret_cast<unsigned*>(p) = pk.u;   // 4B-aligned by construction
}

__device__ __forceinline__ v8f wmma_bf16(v16bf a, v16bf b, v8f c) {
    return __builtin_amdgcn_wmma_f32_16x16x32_bf16(false, a, false, b,
                                                   (short)0, c, false, false);
}

extern "C" __global__ __launch_bounds__(256)
void rollout_wmma_kernel(const float* __restrict__ x,
                         const float* __restrict__ W1, const float* __restrict__ b1,
                         const float* __restrict__ W2, const float* __restrict__ b2,
                         const float* __restrict__ W3, const float* __restrict__ b3,
                         float* __restrict__ out)
{
    extern __shared__ __bf16 smem[];
    __bf16* W1T  = smem;                     // [256][40]   W1^T, K-padded to 32
    __bf16* W2T  = W1T + 256 * W1_STR;       // [256][264]  W2^T (K-rows permuted), dead after preload
    __bf16* W3T  = W2T + 256 * K_STR;        // [16][264]   W3^T (K-rows permuted), N-padded
    __bf16* NET  = W3T + 16  * K_STR;        // [128][40]   net input (per-wave row slices)
    __bf16* HSTa = NET + ROWS * W1_STR;      // [128][264]  h1 stage (perm columns)
    __bf16* HSTb = W2T;                      // [128][264]  h2 stage (perm columns), OVERLAYS W2T

    const int tid  = threadIdx.x;
    const int wave = tid >> 5;
    const int lane = tid & 31;
    const int col  = lane & 15;   // N (and M for A-fetch) within a 16-tile
    const int hi   = lane >> 4;   // lane half -> K sub-range
    const int bO   = hi * 8;      // B-fragment K sub-offset

    // ---- one-time cooperative weight staging (transpose + K-permute) ----
    {
        const int n = tid; // 0..255
        #pragma unroll
        for (int k = 0; k < W1_STR; ++k) W1T[n * W1_STR + k] = (__bf16)0.f;
        for (int f = 0; f < F_IN; ++f)   W1T[n * W1_STR + f] = (__bf16)W1[f * HID + n];
        for (int k = 0; k < HID; ++k)    W2T[n * K_STR + kperm(k)] = (__bf16)W2[k * HID + n];
        if (n < 16) {
            for (int k = 0; k < HID; ++k)
                W3T[n * K_STR + kperm(k)] = (n < NS) ? (__bf16)W3[k * NS + n] : (__bf16)0.f;
        }
    }
    for (int i = tid; i < ROWS * W1_STR; i += 256) NET[i] = (__bf16)0.f;
    __syncthreads();

    // ---- register-resident W2: this wave owns N-tiles {2w, 2w+1} for all rows ----
    v16bf wB[2][8];   // 2 N-tiles x 8 K-chunks = 128 VGPRs, live for whole kernel
    #pragma unroll
    for (int p = 0; p < 2; ++p)
        #pragma unroll
        for (int kc = 0; kc < 8; ++kc)
            wB[p][kc] = ldB(W2T + ((wave * 2 + p) * 16 + col) * K_STR + kc * 32 + bO);
    // make sure every wave's fragments left W2T before anyone overwrites it (HSTb overlay)
    asm volatile("s_wait_dscnt 0x0" ::: "memory");
    __syncthreads();

    // per-lane bias registers
    float b1v[16];
    #pragma unroll
    for (int tI = 0; tI < 16; ++tI) b1v[tI] = b1[tI * 16 + col];
    const float b2v0 = b2[(wave * 2 + 0) * 16 + col];
    const float b2v1 = b2[(wave * 2 + 1) * 16 + col];
    const float b3v  = (col < NS) ? b3[col] : 0.f;

    const int rowbase = blockIdx.x * ROWS + wave * 16;

    // state tile in WMMA C/D layout: VGPR r <-> row r+8*hi, lane col <-> state idx
    v8f st;
    #pragma unroll
    for (int r = 0; r < 8; ++r) {
        int row = rowbase + r + hi * 8;
        st[r] = (col < NS) ? x[(size_t)row * H_T * F_IN + NC + col] : 0.f;
    }

    __bf16* ns = NET + wave * 16 * W1_STR;                         // own net-input rows
    const __bf16* nsA = ns + col * W1_STR + hi * 8;                // layer-1 A base
    const __bf16* hbA = HSTb + (wave * 16 + col) * K_STR + hi * 8; // layer-3 A base (own rows)

    for (int t = 0; t < H_T; ++t) {
        // ---- phase A: build net input [u_t | state] for own rows ----
        if (lane < 16) {
            const float* xp = x + (size_t)(rowbase + lane) * H_T * F_IN + (size_t)t * F_IN;
            ns[lane * W1_STR + 0] = (__bf16)xp[0];
            ns[lane * W1_STR + 1] = (__bf16)xp[1];
        }
        if (col < NS) {
            #pragma unroll
            for (int r = 0; r < 8; ++r)
                ns[(r + hi * 8) * W1_STR + NC + col] = (__bf16)st[r];
        }
        __syncthreads();

        // ---- phase B: layer 1 on own rows, (16x32)@(32x256), tanh -> HSTa (perm cols) ----
        {
            v16bf a1 = ldA(nsA);
            v8f acc[16];
            #pragma unroll
            for (int nt = 0; nt < 16; ++nt) {
                v16bf bm = ldB(W1T + (nt * 16 + col) * W1_STR + bO);
                acc[nt] = wmma_bf16(a1, bm, zero8());
            }
            #pragma unroll
            for (int q = 0; q < 8; ++q)
                #pragma unroll
                for (int r = 0; r < 8; ++r)
                    store_pk(&HSTa[(wave * 16 + r + hi * 8) * K_STR + 32 * q + 2 * col],
                             fast_tanh(acc[2 * q + 0][r] + b1v[2 * q + 0]),
                             fast_tanh(acc[2 * q + 1][r] + b1v[2 * q + 1]));
        }
        __syncthreads();

        // ---- phase C: layer 2, N-split: this wave = cols [32w,32w+32) x all 128 rows.
        //      W2 B-fragments from registers; only A streams from LDS. ----
        #pragma unroll
        for (int rt = 0; rt < 8; ++rt) {
            const __bf16* aB = HSTa + (rt * 16 + col) * K_STR + hi * 8;
            v8f acc0 = zero8(), acc1 = zero8();
            #pragma unroll
            for (int kc = 0; kc < 8; ++kc) {
                v16bf a = ldA(aB + kc * 32);
                acc0 = wmma_bf16(a, wB[0][kc], acc0);
                acc1 = wmma_bf16(a, wB[1][kc], acc1);
            }
            #pragma unroll
            for (int r = 0; r < 8; ++r)
                store_pk(&HSTb[(rt * 16 + r + hi * 8) * K_STR + wave * 32 + 2 * col],
                         fast_tanh(acc0[r] + b2v0),
                         fast_tanh(acc1[r] + b2v1));
        }
        __syncthreads();

        // ---- phase D: layer 3 on own rows, (16x256)@(256x16pad), Euler update ----
        {
            v8f accd = zero8();
            #pragma unroll
            for (int kc = 0; kc < 8; ++kc) {
                v16bf a  = ldA(hbA + kc * 32);
                v16bf bm = ldB(W3T + col * K_STR + kc * 32 + bO);
                accd = wmma_bf16(a, bm, accd);
            }
            #pragma unroll
            for (int r = 0; r < 8; ++r) st[r] += DT_C * (accd[r] + b3v);
            if (col < NS) {
                #pragma unroll
                for (int r = 0; r < 8; ++r)
                    out[(size_t)(rowbase + r + hi * 8) * (H_T * NS) + (size_t)t * NS + col] = st[r];
            }
        }
        // next iteration's phase-A barrier orders HSTb reads vs. next h2 writes
    }
}

extern "C" void kernel_launch(void* const* d_in, const int* in_sizes, int n_in,
                              void* d_out, int out_size, void* d_ws, size_t ws_size,
                              hipStream_t stream)
{
    const float* x  = (const float*)d_in[0];
    const float* W1 = (const float*)d_in[1];
    const float* b1 = (const float*)d_in[2];
    const float* W2 = (const float*)d_in[3];
    const float* b2 = (const float*)d_in[4];
    const float* W3 = (const float*)d_in[5];
    const float* b3 = (const float*)d_in[6];
    float* out = (float*)d_out;

    // W1T + W2T(=HSTb overlay) + W3T + NET + HSTa -> ~236 KB (CDNA5 LDS = 320 KB/WGP)
    const int SMEM = (256 * W1_STR + 256 * K_STR + 16 * K_STR +
                      ROWS * W1_STR + ROWS * K_STR) * 2;
    (void)hipFuncSetAttribute((const void*)rollout_wmma_kernel,
                              hipFuncAttributeMaxDynamicSharedMemorySize, SMEM);

    dim3 grid(B_TOT / ROWS);   // 256 blocks x 128 rows
    rollout_wmma_kernel<<<grid, WAVES * 32, SMEM, stream>>>(x, W1, b1, W2, b2, W3, b3, out);
}
